// MLAAttention_9088150798898
// MI455X (gfx1250) — matrix operand order repaired and checked by reading, not compiled, not run on hardware
//
#include <hip/hip_runtime.h>
#include <stdint.h>

// MI455X / gfx1250 MLA attention, bf16 WMMA path.
// ~160 GFLOP dense vs ~100MB traffic -> compute bound; whole working set fits
// in 192MB L2, so GEMMs load operands directly from global (L2-resident) and
// LDS is used only for the P (probability) C-layout -> A-layout transpose.

#define DEV __device__ __forceinline__

typedef __bf16 v16bf __attribute__((ext_vector_type(16)));
typedef float  v8f   __attribute__((ext_vector_type(8)));

union AFrag { v16bf v; uint4 q[2]; };

DEV uint16_t f2bf(float f) {
  uint32_t u = __float_as_uint(f);
  u += 0x7fffu + ((u >> 16) & 1u);   // round-to-nearest-even
  return (uint16_t)(u >> 16);
}
DEV float bf2f(uint16_t h) { return __uint_as_float(((uint32_t)h) << 16); }

// A-operand fragment (16x32, MxK): lane m(<16) holds row m, K {0..7,16..23};
// lanes 16..31 hold K {8..15,24..31}.  (ISA 7.12.2, 16-bit A 16x32)
DEV void load_fragA(const uint16_t* __restrict__ base, int ld, int row, int k0,
                    int lane, AFrag& f) {
  const uint16_t* p = base + (size_t)(row + (lane & 15)) * ld + k0 + ((lane >> 4) << 3);
  f.q[0] = *(const uint4*)(p);
  f.q[1] = *(const uint4*)(p + 16);
}

// B-operand fragment (32x16, KxN) sourced from row-major W(N,K): lane n(<16)
// holds column n with K {0..7,16..23}; lanes 16..31 hold K {8..15,24..31}.
// Feeding W rows here computes A @ W^T.
DEV void load_fragB(const uint16_t* __restrict__ base, int ld, int row, int k0,
                    int lane, AFrag& f) {
  const uint16_t* p = base + (size_t)(row + (lane & 15)) * ld + k0 + ((lane >> 4) << 3);
  f.q[0] = *(const uint4*)(p);
  f.q[1] = *(const uint4*)(p + 16);
}

DEV v8f wmma_bf16(const AFrag& a, const AFrag& b, v8f c) {
  return __builtin_amdgcn_wmma_f32_16x16x32_bf16(false, a.v, false, b.v,
                                                 (short)0, c, false, false);
}

// ---------------------------------------------------------------- cvt f32->bf16
__global__ void cvt_f32_bf16_kernel(const float* __restrict__ src,
                                    uint16_t* __restrict__ dst, int n) {
  int i = (blockIdx.x * blockDim.x + threadIdx.x) * 4;
  if (i >= n) return;
  float4 v = *(const float4*)(src + i);
  uint2 p;
  p.x = (uint32_t)f2bf(v.x) | ((uint32_t)f2bf(v.y) << 16);
  p.y = (uint32_t)f2bf(v.z) | ((uint32_t)f2bf(v.w) << 16);
  *(uint2*)(dst + i) = p;
}

// ---------------------------------------------------------------- GEMM C=A@W^T
// A:(M,K) bf16, W:(N,K) bf16. Block = 4 waves -> 128x128 tile; wave = 64x64.
// MODE 0: bf16 row-major (ldc=N); MODE 1: bf16 transposed C^T (ldc=M);
// MODE 2: f32 row-major.
template <int MODE>
__global__ __launch_bounds__(128)
void gemm_bt_kernel(const uint16_t* __restrict__ A, const uint16_t* __restrict__ W,
                    void* __restrict__ Cout, int M, int N, int K, int ldc) {
  const int lane = threadIdx.x & 31;
  const int wave = threadIdx.x >> 5;
  const int m0 = blockIdx.y * 128 + (wave >> 1) * 64;
  const int n0 = blockIdx.x * 128 + (wave & 1) * 64;
  (void)M;

  const v8f vzero = {0.f,0.f,0.f,0.f,0.f,0.f,0.f,0.f};
  v8f acc[4][4];
#pragma unroll
  for (int i = 0; i < 4; ++i)
#pragma unroll
    for (int j = 0; j < 4; ++j) acc[i][j] = vzero;

  for (int k0 = 0; k0 < K; k0 += 32) {
    AFrag a[4], b[4];
#pragma unroll
    for (int i = 0; i < 4; ++i) load_fragA(A, K, m0 + i * 16, k0, lane, a[i]);
#pragma unroll
    for (int j = 0; j < 4; ++j) {
      if (n0 + j * 16 < N) load_fragB(W, K, n0 + j * 16, k0, lane, b[j]);
      else { b[j].q[0] = make_uint4(0,0,0,0); b[j].q[1] = make_uint4(0,0,0,0); }
    }
    if (k0 + 32 < K)  // warm L2/L0 for next K-slab (global_prefetch_b8)
      __builtin_prefetch(A + (size_t)(m0 + (lane & 15)) * K + k0 + 32, 0, 1);
#pragma unroll
    for (int i = 0; i < 4; ++i)
#pragma unroll
      for (int j = 0; j < 4; ++j)
        acc[i][j] = wmma_bf16(a[i], b[j], acc[i][j]);
  }

  // C/D layout: VGPR r -> row m0+i*16+r (+8 for upper half-wave), col = lane&15
  const int mbase = (lane >> 4) << 3;
  const int ncol  = lane & 15;
#pragma unroll
  for (int i = 0; i < 4; ++i)
#pragma unroll
    for (int j = 0; j < 4; ++j) {
      const int n = n0 + j * 16 + ncol;
      if (n >= N) continue;
#pragma unroll
      for (int r = 0; r < 8; ++r) {
        const int m = m0 + i * 16 + mbase + r;
        const float v = acc[i][j][r];
        if (MODE == 0)      ((uint16_t*)Cout)[(size_t)m * ldc + n] = f2bf(v);
        else if (MODE == 1) ((uint16_t*)Cout)[(size_t)n * ldc + m] = f2bf(v);
        else                ((float*)Cout)[(size_t)m * ldc + n]    = v;
      }
    }
}

// ---------------------------------------------------------------- RoPE (in place)
// q: (B*S, 2048) bf16, rope half = cols [h*128+64, h*128+128); kr: (B*S, 64).
// unit 0..543: 16 heads * 32 pairs + 32 pairs for k_rope (h==16).
__global__ __launch_bounds__(544)
void rope_kernel(uint16_t* __restrict__ qb, uint16_t* __restrict__ kr) {
  const int row  = blockIdx.x;            // b*S + s
  const int pos  = row & 2047;            // s (S = 2048)
  const int unit = threadIdx.x;
  const int hh = unit >> 5;
  const int i  = unit & 31;
  const float freq = __powf(10000.0f, -((float)(2 * i)) * (1.0f / 64.0f));
  float sn, cs;
  __sincosf((float)pos * freq, &sn, &cs);
  uint16_t* p = (hh < 16)
      ? (qb + (size_t)row * 2048 + hh * 128 + 64 + 2 * i)
      : (kr + (size_t)row * 64 + 2 * i);
  const float x0 = bf2f(p[0]);
  const float x1 = bf2f(p[1]);
  p[0] = f2bf(x0 * cs - x1 * sn);
  p[1] = f2bf(x1 * cs + x0 * sn);
}

// ---------------------------------------------------------------- flash attention
// grid.x = b*16+h, grid.y = S/128; 8 waves/block, each wave owns 16 q rows.
// Per 32-wide k block: 8 wmma (scores, K-dim 128 = nope64|rope64), online
// softmax with 16-lane shfl reductions, P transposed via 1KB wave-private LDS,
// 8 wmma for P @ V (V pre-transposed: vT[(h*128+d)*4096 + b*2048 + s]).
// All fixed-trip loops force-unrolled so per-row stats stay in static VGPRs
// (no M0/v_movrel dynamic indexing).
__global__ __launch_bounds__(256)
void mla_attn_kernel(const uint16_t* __restrict__ q,
                     const uint16_t* __restrict__ knope,
                     const uint16_t* __restrict__ krope,
                     const uint16_t* __restrict__ vT,
                     uint16_t* __restrict__ out) {
  constexpr int S = 2048;
  __shared__ __align__(16) uint16_t pbuf[8 * 16 * 32];

  const int lane = threadIdx.x & 31;
  const int wave = threadIdx.x >> 5;
  const int b = blockIdx.x >> 4;
  const int h = blockIdx.x & 15;
  const int qr0 = blockIdx.y * 128 + wave * 16;

  const uint16_t* qbase = q     + (size_t)b * S * 2048 + h * 128;   // ld 2048
  const uint16_t* knb   = knope + (size_t)b * S * 1024 + h * 64;    // ld 1024
  const uint16_t* krb   = krope + (size_t)b * S * 64;               // ld 64
  const uint16_t* vtb   = vT    + (size_t)h * 128 * 4096 + b * S;   // ld 4096
  uint16_t* myp = pbuf + wave * 512;

  AFrag af[4];                       // Q rows, d = 0..127 (nope|rope contiguous)
#pragma unroll
  for (int t = 0; t < 4; ++t) load_fragA(qbase, 2048, qr0, t * 32, lane, af[t]);

  const v8f vzero = {0.f,0.f,0.f,0.f,0.f,0.f,0.f,0.f};
  v8f o[8];
#pragma unroll
  for (int t = 0; t < 8; ++t) o[t] = vzero;
  float mrow[8], lrow[8];
#pragma unroll
  for (int r = 0; r < 8; ++r) { mrow[r] = -1e30f; lrow[r] = 0.0f; }

  const int mbase = (lane >> 4) << 3;
  const int ncol  = lane & 15;
  const float scale = 0.08838834764831845f;   // 1/sqrt(128)
  const int kend = qr0 + 16;                  // causal horizon (exclusive)

  for (int k0 = 0; k0 < kend; k0 += 32) {
    // ---- scores: issue all 8 B-fragment loads, then 8 WMMAs
    AFrag bn[2][2], br[2][2];
#pragma unroll
    for (int j = 0; j < 2; ++j) {
      load_fragB(knb, 1024, k0 + j * 16, 0,  lane, bn[j][0]);
      load_fragB(knb, 1024, k0 + j * 16, 32, lane, bn[j][1]);
      load_fragB(krb,   64, k0 + j * 16, 0,  lane, br[j][0]);
      load_fragB(krb,   64, k0 + j * 16, 32, lane, br[j][1]);
    }
    v8f sc[2];
#pragma unroll
    for (int j = 0; j < 2; ++j) {
      sc[j] = vzero;
      sc[j] = wmma_bf16(af[0], bn[j][0], sc[j]);   // q_nope . k_nope
      sc[j] = wmma_bf16(af[1], bn[j][1], sc[j]);
      sc[j] = wmma_bf16(af[2], br[j][0], sc[j]);   // q_rope . k_rope
      sc[j] = wmma_bf16(af[3], br[j][1], sc[j]);
    }
    // ---- scale, causal mask, online softmax update (rows mbase..mbase+7)
#pragma unroll
    for (int r = 0; r < 8; ++r) {
      const int qs = qr0 + mbase + r;
      float s0 = sc[0][r] * scale; if (k0 + ncol      > qs) s0 = -1e30f;
      float s1 = sc[1][r] * scale; if (k0 + 16 + ncol > qs) s1 = -1e30f;
      float mx = fmaxf(s0, s1);
      mx = fmaxf(mx, __shfl_xor(mx, 1));
      mx = fmaxf(mx, __shfl_xor(mx, 2));
      mx = fmaxf(mx, __shfl_xor(mx, 4));
      mx = fmaxf(mx, __shfl_xor(mx, 8));      // row max within 16-lane group
      const float mnew = fmaxf(mrow[r], mx);
      const float p0 = __expf(s0 - mnew);
      const float p1 = __expf(s1 - mnew);
      float rs = p0 + p1;
      rs += __shfl_xor(rs, 1);
      rs += __shfl_xor(rs, 2);
      rs += __shfl_xor(rs, 4);
      rs += __shfl_xor(rs, 8);
      const float corr = __expf(mrow[r] - mnew);
      lrow[r] = lrow[r] * corr + rs;
      mrow[r] = mnew;
#pragma unroll
      for (int t = 0; t < 8; ++t) o[t][r] *= corr;
      myp[(mbase + r) * 32 + ncol]      = f2bf(p0);   // P row-major in LDS
      myp[(mbase + r) * 32 + 16 + ncol] = f2bf(p1);
    }
    asm volatile("s_wait_dscnt 0" ::: "memory");      // LDS in-order per wave
    AFrag pf;                                         // re-read in A-frag layout
    {
      const uint16_t* pp = myp + (lane & 15) * 32 + ((lane >> 4) << 3);
      pf.q[0] = *(const uint4*)(pp);
      pf.q[1] = *(const uint4*)(pp + 16);
    }
    // ---- o += P @ V, pairs of V^T fragments in flight
#pragma unroll
    for (int t = 0; t < 8; t += 2) {
      AFrag bv0, bv1;
      load_fragB(vtb, 4096, t * 16,       k0, lane, bv0);
      load_fragB(vtb, 4096, (t + 1) * 16, k0, lane, bv1);
      o[t]     = wmma_bf16(pf, bv0, o[t]);
      o[t + 1] = wmma_bf16(pf, bv1, o[t + 1]);
    }
  }

#pragma unroll
  for (int r = 0; r < 8; ++r) {
    const int qs = qr0 + mbase + r;
    const float inv = 1.0f / lrow[r];
#pragma unroll
    for (int t = 0; t < 8; ++t) {
      const int d = t * 16 + ncol;
      out[(size_t)(b * S + qs) * 2048 + h * 128 + d] = f2bf(o[t][r] * inv);
    }
  }
}

// ---------------------------------------------------------------- launch
extern "C" void kernel_launch(void* const* d_in, const int* in_sizes, int n_in,
                              void* d_out, int out_size, void* d_ws, size_t ws_size,
                              hipStream_t stream) {
  (void)in_sizes; (void)n_in; (void)out_size; (void)ws_size;
  const float* x         = (const float*)d_in[0];
  const float* wq        = (const float*)d_in[1];
  const float* w_kv_down = (const float*)d_in[2];
  const float* w_k_up    = (const float*)d_in[3];
  const float* w_k_rope  = (const float*)d_in[4];
  const float* w_v_up    = (const float*)d_in[5];
  const float* wo        = (const float*)d_in[6];

  const int BS = 4096, D = 2048, R = 512, HND = 1024, RD = 64;

  char* ws = (char*)d_ws;
  size_t off = 0;
  auto take = [&](size_t elems) {
    uint16_t* p = (uint16_t*)(ws + off);
    off += elems * sizeof(uint16_t);
    return p;
  };
  uint16_t* xb   = take((size_t)BS * D);     // x bf16
  uint16_t* wqb  = take((size_t)D * D);
  uint16_t* wkvb = take((size_t)R * D);
  uint16_t* wkub = take((size_t)HND * R);
  uint16_t* wkrb = take((size_t)RD * D);
  uint16_t* wvub = take((size_t)D * R);
  uint16_t* wob  = take((size_t)D * D);
  uint16_t* qb   = take((size_t)BS * D);     // q (rope applied in place)
  uint16_t* ckv  = take((size_t)BS * R);
  uint16_t* knp  = take((size_t)BS * HND);
  uint16_t* krp  = take((size_t)BS * RD);
  uint16_t* vTb  = take((size_t)D * BS);     // v stored transposed
  uint16_t* ao   = take((size_t)BS * D);     // attention output

  auto cvt = [&](const float* s, uint16_t* d, size_t n) {
    int blocks = (int)((n / 4 + 255) / 256);
    cvt_f32_bf16_kernel<<<blocks, 256, 0, stream>>>(s, d, (int)n);
  };
  cvt(x,         xb,   (size_t)BS * D);
  cvt(wq,        wqb,  (size_t)D * D);
  cvt(w_kv_down, wkvb, (size_t)R * D);
  cvt(w_k_up,    wkub, (size_t)HND * R);
  cvt(w_k_rope,  wkrb, (size_t)RD * D);
  cvt(w_v_up,    wvub, (size_t)D * R);
  cvt(wo,        wob,  (size_t)D * D);

  const dim3 blk(128);
  // q = x @ wq^T
  gemm_bt_kernel<0><<<dim3(D / 128, BS / 128), blk, 0, stream>>>(xb, wqb, qb, BS, D, D, D);
  // c_kv = x @ w_kv_down^T
  gemm_bt_kernel<0><<<dim3(R / 128, BS / 128), blk, 0, stream>>>(xb, wkvb, ckv, BS, R, D, R);
  // k_rope_pre = x @ w_k_rope^T (N=64 partial tile, guarded)
  gemm_bt_kernel<0><<<dim3(1, BS / 128), blk, 0, stream>>>(xb, wkrb, krp, BS, RD, D, RD);
  // k_nope = c_kv @ w_k_up^T
  gemm_bt_kernel<0><<<dim3(HND / 128, BS / 128), blk, 0, stream>>>(ckv, wkub, knp, BS, HND, R, HND);
  // v^T = (c_kv @ w_v_up^T)^T
  gemm_bt_kernel<1><<<dim3(D / 128, BS / 128), blk, 0, stream>>>(ckv, wvub, vTb, BS, D, R, BS);
  // RoPE on q rope-halves and k_rope
  rope_kernel<<<BS, 544, 0, stream>>>(qb, krp);
  // flash attention
  mla_attn_kernel<<<dim3(32, 16), 256, 0, stream>>>(qb, knp, krp, vTb, ao);
  // out = attn_out @ wo^T (f32)
  gemm_bt_kernel<2><<<dim3(D / 128, BS / 128), blk, 0, stream>>>(ao, wob, d_out, BS, D, D, D);
}
